// LipNet_91173565759852
// MI455X (gfx1250) — matrix operand order repaired and checked by reading, not compiled, not run on hardware
//
#include <hip/hip_runtime.h>
#include <hip/hip_bf16.h>
#include <math.h>

// ---------------------------------------------------------------------------
// CDNA5 (gfx1250) LipNet forward pipeline.
// All GEMM-shaped work uses v_wmma_f32_16x16x32_bf16 (wave32, fp32 accum).
// Round 2: branchless clamped staging (no exec-mask branches in the k-loop),
// 64x128 GEMM tiles (4 WMMA / wave / k-step), double-buffered LDS with a
// single barrier per k-step, prefetch two tiles ahead.
// ---------------------------------------------------------------------------

typedef __bf16 bf16_t;
typedef __attribute__((ext_vector_type(8)))  __bf16 v8bf;
typedef __attribute__((ext_vector_type(16))) __bf16 v16bf;
typedef __attribute__((ext_vector_type(8)))  float  v8f;

union FragBF { v16bf v; v8bf h[2]; };

__device__ __forceinline__ v8f wmma_bf16f32(const FragBF& a, const FragBF& b, v8f c) {
  // 8 args: (neg_a, A, neg_b, B, c_mod, C, reuse_a, reuse_b)
  return __builtin_amdgcn_wmma_f32_16x16x32_bf16(false, a.v, false, b.v, (short)0, c,
                                                 false, false);
}

__device__ __forceinline__ float sig_f(float x) { return 1.f / (1.f + expf(-x)); }

// ---------------------------------------------------------------------------
// Generic GEMM: C[M,N] = A[M,K](bf16,row-major) x W[N,K](bf16,row-major)^T
//  + bias0 (+bias1), optional relu. Block = 256 threads = 8 waves.
// Block tile 64(M) x 128(N); each wave 16x64 via 4 accumulators; K mult of 32.
// OOB M/N rows are clamped (their C entries are never stored), so staging is
// branchless. LDS ping-pong => one barrier per k-step.
// ---------------------------------------------------------------------------
__global__ __launch_bounds__(256)
void gemm_bf16_kernel(const bf16_t* __restrict__ A, const bf16_t* __restrict__ W,
                      const float* __restrict__ bias0, const float* __restrict__ bias1,
                      float* __restrict__ Cf, bf16_t* __restrict__ Cb,
                      int M, int N, int K, int relu)
{
  __shared__ __align__(16) bf16_t sA[2][64][32];
  __shared__ __align__(16) bf16_t sB[2][128][32];
  const int tid  = threadIdx.x;
  const int wave = tid >> 5, lane = tid & 31;
  const int hf   = lane >> 4, lr = lane & 15;
  const int m0 = blockIdx.x * 64, n0 = blockIdx.y * 128;
  const int wm = (wave & 3) * 16, wn = (wave >> 2) * 64;
  const int sr = tid >> 2, sc = (tid & 3) * 8;

  // clamped source rows: OOB rows load duplicate data, masked at store
  int ra  = m0 + sr;      if (ra  >= M) ra  = M - 1;
  int rb0 = n0 + sr;      if (rb0 >= N) rb0 = N - 1;
  int rb1 = n0 + sr + 64; if (rb1 >= N) rb1 = N - 1;
  const bf16_t* pa  = A + (size_t)ra  * K + sc;
  const bf16_t* pw0 = W + (size_t)rb0 * K + sc;
  const bf16_t* pw1 = W + (size_t)rb1 * K + sc;

  auto stage = [&](int p, int k0) {
    *(v8bf*)&sA[p][sr][sc]      = *(const v8bf*)(pa  + k0);
    *(v8bf*)&sB[p][sr][sc]      = *(const v8bf*)(pw0 + k0);
    *(v8bf*)&sB[p][sr + 64][sc] = *(const v8bf*)(pw1 + k0);
  };

  v8f acc[4] = {{}, {}, {}, {}};
  const int nk = K >> 5;
  stage(0, 0);
  for (int s = 0; s < nk; ++s) {
    __syncthreads();
    if (s + 1 < nk) {
      stage((s + 1) & 1, (s + 1) << 5);
      if (s + 2 < nk) {  // global_prefetch_b8 two tiles ahead
        __builtin_prefetch(pa  + ((s + 2) << 5), 0, 1);
        __builtin_prefetch(pw0 + ((s + 2) << 5), 0, 1);
        __builtin_prefetch(pw1 + ((s + 2) << 5), 0, 1);
      }
    }
    const int p = s & 1;
    FragBF fa;
    fa.h[0] = *(const v8bf*)&sA[p][wm + lr][hf * 8];
    fa.h[1] = *(const v8bf*)&sA[p][wm + lr][16 + hf * 8];
    #pragma unroll
    for (int j = 0; j < 4; ++j) {
      FragBF fb;
      fb.h[0] = *(const v8bf*)&sB[p][wn + j * 16 + lr][hf * 8];
      fb.h[1] = *(const v8bf*)&sB[p][wn + j * 16 + lr][16 + hf * 8];
      acc[j] = wmma_bf16f32(fa, fb, acc[j]);
    }
  }
  // C layout: vgpr r -> row = r + 8*hf, col = lr (per 16x16 tile)
  #pragma unroll
  for (int r = 0; r < 8; ++r) {
    int row = m0 + wm + hf * 8 + r;
    if (row >= M) continue;
    #pragma unroll
    for (int j = 0; j < 4; ++j) {
      int col = n0 + wn + j * 16 + lr;
      if (col < N) {
        float v = acc[j][r] + bias0[col];
        if (bias1) v += bias1[col];
        if (relu)  v = fmaxf(v, 0.f);
        size_t o = (size_t)row * N + col;
        if (Cf) Cf[o] = v;
        if (Cb) Cb[o] = (bf16_t)v;
      }
    }
  }
}

// ---------------------------------------------------------------------------
// Implicit-GEMM 3x3x3 conv (stride 1, pad (1,pad,pad)), channels-last bf16.
// act:[B*T,Hin,Win,Cin]  wt:[27][Cout][Cin]  out:[B*T,Hout,Wout,Cout] (relu).
// Flattened (tap, k-chunk) loop, double-buffered LDS, branchless staging
// (clamped coords + vector select for zero padding).
// ---------------------------------------------------------------------------
__global__ __launch_bounds__(256)
void conv3d_gemm_kernel(const bf16_t* __restrict__ act, const bf16_t* __restrict__ wt,
                        const float* __restrict__ bias, bf16_t* __restrict__ out,
                        int Mtot, int Hin, int Win, int Cin,
                        int Hout, int Wout, int Cout, int pad)
{
  __shared__ __align__(16) bf16_t sA[2][64][32];
  __shared__ __align__(16) bf16_t sB[2][64][32];
  __shared__ int rB[64], rT[64], rH[64], rW[64], rV[64];
  const int tid  = threadIdx.x;
  const int wave = tid >> 5, lane = tid & 31;
  const int hf   = lane >> 4, lr = lane & 15;
  const int m0 = blockIdx.x * 64, n0 = blockIdx.y * 64;
  const int wm = (wave & 3) * 16, wn = (wave >> 2) * 32;
  const int sr = tid >> 2, sc = (tid & 3) * 8;

  if (tid < 64) {
    int p = m0 + tid;
    int valid = p < Mtot;
    int pp = valid ? p : 0;           // clamp decode; masked at store
    int w  = pp % Wout;
    int h  = (pp / Wout) % Hout;
    int bt = pp / (Wout * Hout);
    rB[tid] = bt / 75; rT[tid] = bt % 75; rH[tid] = h; rW[tid] = w; rV[tid] = valid;
  }
  __syncthreads();

  const int aB = rB[sr], aT = rT[sr], aH = rH[sr], aW = rW[sr];
  int nB = n0 + sr; if (nB >= Cout) nB = Cout - 1;   // clamp W row

  const int kcn = Cin >> 5;     // k-chunks per tap (1 or 2)
  const int nk  = 27 * kcn;

  auto stage = [&](int p, int s) {
    int tap = s / kcn;
    int kc  = (s - tap * kcn) << 5;
    int kd = tap / 9, kh = (tap / 3) % 3, kw = tap % 3;
    int d = aT + kd - 1, y = aH + kh - pad, x = aW + kw - pad;
    bool valid = (d >= 0) && (d < 75) && (y >= 0) && (y < Hin) && (x >= 0) && (x < Win);
    int dc = d < 0 ? 0 : (d > 74 ? 74 : d);
    int yc = y < 0 ? 0 : (y >= Hin ? Hin - 1 : y);
    int xc = x < 0 ? 0 : (x >= Win ? Win - 1 : x);
    size_t base = ((((size_t)aB * 75 + dc) * Hin + yc) * Win + xc) * Cin + kc + sc;
    v8bf av = *(const v8bf*)(act + base);
    v8bf zv = {};
    av = valid ? av : zv;            // zero padding, branchless
    *(v8bf*)&sA[p][sr][sc] = av;
    *(v8bf*)&sB[p][sr][sc] = *(const v8bf*)(wt + ((size_t)tap * Cout + nB) * Cin + kc + sc);
  };

  v8f acc0 = {}; v8f acc1 = {};
  stage(0, 0);
  for (int s = 0; s < nk; ++s) {
    __syncthreads();
    if (s + 1 < nk) stage((s + 1) & 1, s + 1);
    const int p = s & 1;
    FragBF fa, fb0, fb1;
    fa.h[0]  = *(const v8bf*)&sA[p][wm + lr][hf * 8];
    fa.h[1]  = *(const v8bf*)&sA[p][wm + lr][16 + hf * 8];
    fb0.h[0] = *(const v8bf*)&sB[p][wn + lr][hf * 8];
    fb0.h[1] = *(const v8bf*)&sB[p][wn + lr][16 + hf * 8];
    fb1.h[0] = *(const v8bf*)&sB[p][wn + 16 + lr][hf * 8];
    fb1.h[1] = *(const v8bf*)&sB[p][wn + 16 + lr][16 + hf * 8];
    acc0 = wmma_bf16f32(fa, fb0, acc0);
    acc1 = wmma_bf16f32(fa, fb1, acc1);
  }
  #pragma unroll
  for (int r = 0; r < 8; ++r) {
    int ri = wm + hf * 8 + r;
    if (!rV[ri]) continue;
    size_t pix = (((size_t)rB[ri] * 75 + rT[ri]) * Hout + rH[ri]) * Wout + rW[ri];
    #pragma unroll
    for (int t = 0; t < 2; ++t) {
      int n = n0 + wn + t * 16 + lr;
      if (n < Cout) {
        float v = (t == 0 ? acc0[r] : acc1[r]) + bias[n];
        out[pix * Cout + n] = (bf16_t)fmaxf(v, 0.f);
      }
    }
  }
}

// ---------------------------------------------------------------------------
// conv1 (Cin=1) + bias + relu + maxpool(1,2,2) fused. VALU fp32.
// x:[16,75,64,128] fp32 -> out:[16,75,16,32,32] bf16 channels-last.
// ---------------------------------------------------------------------------
__global__ __launch_bounds__(256)
void conv1_fused_kernel(const float* __restrict__ x, const float* __restrict__ w,
                        const float* __restrict__ bias, bf16_t* __restrict__ out)
{
  __shared__ float sW[864];
  __shared__ float sBias[32];
  int tid = threadIdx.x;
  for (int i = tid; i < 864; i += 256) sW[i] = w[i];
  if (tid < 32) sBias[tid] = bias[tid];
  __syncthreads();

  long long idx = (long long)blockIdx.x * 256 + tid;   // total 19,660,800
  int c  = (int)(idx & 31);
  int pw = (int)((idx >> 5) & 31);
  int ph = (int)((idx >> 10) & 15);
  long long r = idx >> 14;
  int t = (int)(r % 75);
  int b = (int)(r / 75);
  if (b >= 16) return;

  const float* wc = &sW[c * 27];
  float m = -3.4e38f;
  for (int hh = 0; hh < 2; ++hh) {
    int h = 2 * ph + hh;
    for (int ww = 0; ww < 2; ++ww) {
      int wv = 2 * pw + ww;
      float acc = 0.f;
      #pragma unroll
      for (int kd = 0; kd < 3; ++kd) {
        int d = t + kd - 1;
        if (d < 0 || d >= 75) continue;
        #pragma unroll
        for (int kh = 0; kh < 3; ++kh) {
          int y = 2 * h + kh - 2;
          if (y < 0 || y >= 64) continue;
          #pragma unroll
          for (int kw = 0; kw < 3; ++kw) {
            int xx = 2 * wv + kw - 2;
            if (xx < 0 || xx >= 128) continue;
            acc += x[(((size_t)b * 75 + d) * 64 + y) * 128 + xx] * wc[kd * 9 + kh * 3 + kw];
          }
        }
      }
      m = fmaxf(m, acc);
    }
  }
  out[idx] = (bf16_t)fmaxf(m + sBias[c], 0.f);
}

// pool2: [1200,18,34,64] -> [1200,9,17,64]
__global__ void pool2_kernel(const bf16_t* __restrict__ in, bf16_t* __restrict__ out)
{
  long long idx = (long long)blockIdx.x * 256 + threadIdx.x;
  const long long total = 1200LL * 9 * 17 * 64;
  if (idx >= total) return;
  int c = (int)(idx & 63);
  long long q = idx >> 6;
  int w = (int)(q % 17); q /= 17;
  int h = (int)(q % 9);  q /= 9;
  size_t base = ((q * 18 + 2 * h) * 34 + 2 * w) * 64 + c;
  float v0 = (float)in[base];
  float v1 = (float)in[base + 64];
  float v2 = (float)in[base + 34 * 64];
  float v3 = (float)in[base + 34 * 64 + 64];
  out[idx] = (bf16_t)fmaxf(fmaxf(v0, v1), fmaxf(v2, v3));
}

// pool3 + flatten: [1200,9,17,96] -> xflat [1200, 3072], feature = c*32 + h*8 + w
__global__ void pool3_flat_kernel(const bf16_t* __restrict__ in, bf16_t* __restrict__ out)
{
  long long idx = (long long)blockIdx.x * 256 + threadIdx.x;
  const long long total = 1200LL * 96 * 4 * 8;
  if (idx >= total) return;
  int w = (int)(idx & 7);
  int h = (int)((idx >> 3) & 3);
  long long q = idx >> 5;
  int c = (int)(q % 96);
  long long bt = q / 96;
  size_t base = ((bt * 9 + 2 * h) * 17 + 2 * w) * 96 + c;
  float v0 = (float)in[base], v1 = (float)in[base + 96];
  float v2 = (float)in[base + 17 * 96], v3 = (float)in[base + 17 * 96 + 96];
  out[bt * 3072 + c * 32 + h * 8 + w] = (bf16_t)fmaxf(fmaxf(v0, v1), fmaxf(v2, v3));
}

// delta features: x fp32 [1200,512] -> feats bf16 [1200,1536] = [x | d1 | d2]
__global__ void delta_feats_kernel(const float* __restrict__ x, bf16_t* __restrict__ feats)
{
  int idx = blockIdx.x * 256 + threadIdx.x;
  if (idx >= 1200 * 512) return;
  int f = idx & 511;
  int bt = idx >> 9;
  int b = bt / 75, t = bt % 75;
  const float* xb = x + (size_t)b * 75 * 512 + f;
  auto X = [&](int tt) -> float {
    tt = tt < 0 ? 0 : (tt > 74 ? 74 : tt);
    return xb[(size_t)tt * 512];
  };
  auto D1 = [&](int tt) -> float {
    tt = tt < 0 ? 0 : (tt > 74 ? 74 : tt);
    return (X(tt + 1) - X(tt - 1)) * 0.5f + (X(tt + 2) - X(tt - 2)) * 0.25f;
  };
  float xv = X(t);
  float d1 = D1(t);
  float d2 = (D1(t + 1) - D1(t - 1)) * 0.5f + (D1(t + 2) - D1(t - 2)) * 0.25f;
  size_t o = (size_t)bt * 1536;
  feats[o + f]        = (bf16_t)xv;
  feats[o + 512 + f]  = (bf16_t)d1;
  feats[o + 1024 + f] = (bf16_t)d2;
}

// ---------------------------------------------------------------------------
// LSTM recurrence: single workgroup (1024 threads = 32 waves), 75 steps.
// Per step: g[16,1024] = gates_in[:,t,:] + h(16x256) x w_hh^T (WMMA, w_hh L2
// resident), then elementwise cell update. h/c in registers, hbf16 + gate
// buffer in dynamic LDS (73728 B).
// ---------------------------------------------------------------------------
__global__ __launch_bounds__(1024)
void lstm_kernel(const float* __restrict__ gates, const bf16_t* __restrict__ whh,
                 const float* __restrict__ h0, const float* __restrict__ c0,
                 float* __restrict__ z, float* __restrict__ hN, float* __restrict__ cN)
{
  extern __shared__ char smem[];
  float*  gbuf = (float*)smem;                      // 16*1024 fp32 = 64 KB
  bf16_t* hb   = (bf16_t*)(smem + 16 * 1024 * 4);   // 16*256 bf16 =  8 KB
  const int tid  = threadIdx.x;
  const int wave = tid >> 5, lane = tid & 31;
  const int hf = lane >> 4, lr = lane & 15;

  float hreg[4], creg[4];
  #pragma unroll
  for (int k = 0; k < 4; ++k) {
    int idx = tid + k * 1024;
    int m = idx >> 8, j = idx & 255;
    hreg[k] = h0[m * 256 + j];
    creg[k] = c0[m * 256 + j];
    hb[m * 256 + j] = (bf16_t)hreg[k];
  }
  __syncthreads();

  const int n0 = wave * 32;
  for (int t = 0; t < 75; ++t) {
    #pragma unroll
    for (int tt = 0; tt < 2; ++tt) {
      int nb = n0 + tt * 16;
      v8f acc = {};
      #pragma unroll
      for (int k0 = 0; k0 < 256; k0 += 32) {
        FragBF fa, fb;
        fa.h[0] = *(const v8bf*)&hb[lr * 256 + k0 + hf * 8];
        fa.h[1] = *(const v8bf*)&hb[lr * 256 + k0 + 16 + hf * 8];
        const bf16_t* wrow = whh + (size_t)(nb + lr) * 256 + k0 + hf * 8;
        fb.h[0] = *(const v8bf*)wrow;
        fb.h[1] = *(const v8bf*)(wrow + 16);
        acc = wmma_bf16f32(fa, fb, acc);
      }
      #pragma unroll
      for (int r = 0; r < 8; ++r) {
        int m = hf * 8 + r;
        int n = nb + lr;
        gbuf[m * 1024 + n] = acc[r] + gates[((size_t)m * 75 + t) * 1024 + n];
      }
    }
    __syncthreads();
    #pragma unroll
    for (int k = 0; k < 4; ++k) {
      int idx = tid + k * 1024;
      int m = idx >> 8, j = idx & 255;
      float gi = gbuf[m * 1024 + j];
      float gf = gbuf[m * 1024 + 256 + j];
      float gg = gbuf[m * 1024 + 512 + j];
      float go = gbuf[m * 1024 + 768 + j];
      float cc = sig_f(gf) * creg[k] + sig_f(gi) * tanhf(gg);
      float hh = sig_f(go) * tanhf(cc);
      creg[k] = cc; hreg[k] = hh;
      hb[m * 256 + j] = (bf16_t)hh;
      z[((size_t)m * 75 + t) * 256 + j] = hh;
    }
    __syncthreads();
  }
  #pragma unroll
  for (int k = 0; k < 4; ++k) {
    int idx = tid + k * 1024;
    int m = idx >> 8, j = idx & 255;
    hN[m * 256 + j] = hreg[k];
    cN[m * 256 + j] = creg[k];
  }
}

// softmax over 500 classes; one wave32 per row.
__global__ void softmax_kernel(const float* __restrict__ logits, float* __restrict__ probs)
{
  int row = blockIdx.x * 8 + (threadIdx.x >> 5);
  int lane = threadIdx.x & 31;
  if (row >= 1200) return;
  const float* L = logits + (size_t)row * 500;
  float* P = probs + (size_t)row * 500;
  float m = -3.4e38f;
  for (int j = lane; j < 500; j += 32) m = fmaxf(m, L[j]);
  for (int o = 16; o; o >>= 1) m = fmaxf(m, __shfl_xor(m, o, 32));
  float s = 0.f;
  for (int j = lane; j < 500; j += 32) { float e = expf(L[j] - m); P[j] = e; s += e; }
  for (int o = 16; o; o >>= 1) s += __shfl_xor(s, o, 32);
  float inv = 1.f / s;
  for (int j = lane; j < 500; j += 32) P[j] *= inv;
}

// fp32 -> bf16 elementwise
__global__ void f2bf_kernel(const float* __restrict__ in, bf16_t* __restrict__ out, int n)
{
  int i = blockIdx.x * 256 + threadIdx.x;
  if (i < n) out[i] = (bf16_t)in[i];
}

// conv weight permute+convert: [O,I,3,3,3] fp32 -> [27][O][I] bf16
__global__ void cvt_tap_kernel(const float* __restrict__ in, bf16_t* __restrict__ out,
                               int Cout, int Cin)
{
  int idx = blockIdx.x * 256 + threadIdx.x;
  int total = 27 * Cout * Cin;
  if (idx >= total) return;
  int i = idx % Cin;
  int o = (idx / Cin) % Cout;
  int j = idx / (Cin * Cout);
  out[idx] = (bf16_t)in[(o * Cin + i) * 27 + j];
}

// ---------------------------------------------------------------------------
extern "C" void kernel_launch(void* const* d_in, const int* in_sizes, int n_in,
                              void* d_out, int out_size, void* d_ws, size_t ws_size,
                              hipStream_t stream)
{
  (void)in_sizes; (void)n_in; (void)out_size; (void)ws_size;
  const float* x       = (const float*)d_in[0];
  const float* h0      = (const float*)d_in[1];
  const float* c0      = (const float*)d_in[2];
  const float* conv1_w = (const float*)d_in[3];
  const float* conv1_b = (const float*)d_in[4];
  const float* conv2_w = (const float*)d_in[5];
  const float* conv2_b = (const float*)d_in[6];
  const float* conv3_w = (const float*)d_in[7];
  const float* conv3_b = (const float*)d_in[8];
  const float* lin1_w  = (const float*)d_in[9];
  const float* lin1_b  = (const float*)d_in[10];
  const float* lin2_w  = (const float*)d_in[11];
  const float* lin2_b  = (const float*)d_in[12];
  const float* w_ih    = (const float*)d_in[13];
  const float* w_hh    = (const float*)d_in[14];
  const float* b_ih    = (const float*)d_in[15];
  const float* b_hh    = (const float*)d_in[16];
  const float* cls_w   = (const float*)d_in[17];
  const float* cls_b   = (const float*)d_in[18];
  float* out = (float*)d_out;

  // ---- workspace arenas (bump-allocated, 256B aligned) --------------------
  char* ws = (char*)d_ws;
  size_t off = 0;
  auto alloc = [&](size_t bytes) -> void* {
    void* p = ws + off;
    off += (bytes + 255) & ~(size_t)255;
    return p;
  };
  bf16_t* wt2   = (bf16_t*)alloc((size_t)27 * 64 * 32 * 2);
  bf16_t* wt3   = (bf16_t*)alloc((size_t)27 * 96 * 64 * 2);
  bf16_t* l1w   = (bf16_t*)alloc((size_t)2048 * 3072 * 2);
  bf16_t* l2w   = (bf16_t*)alloc((size_t)512 * 2048 * 2);
  bf16_t* wihb  = (bf16_t*)alloc((size_t)1024 * 1536 * 2);
  bf16_t* whhb  = (bf16_t*)alloc((size_t)1024 * 256 * 2);
  bf16_t* clswb = (bf16_t*)alloc((size_t)500 * 256 * 2);
  char* bufA = (char*)alloc(39321600);  // peak: act1 [16,75,16,32,32] bf16
  char* bufB = (char*)alloc(94003200);  // peak: act2 [1200,18,34,64] bf16

  bf16_t* act1  = (bf16_t*)bufA;   // conv1 out
  bf16_t* act2  = (bf16_t*)bufB;   // conv2 out (pre-pool)
  bf16_t* act2p = (bf16_t*)bufA;   // pool2 out (act1 dead)
  bf16_t* act3  = (bf16_t*)bufB;   // conv3 out (act2 dead)
  bf16_t* xflat = (bf16_t*)bufA;   // pool3+flatten [1200,3072]
  bf16_t* a1    = (bf16_t*)bufB;   // lin1 out [1200,2048]
  float*  x512  = (float*)bufA;    // lin2 out [1200,512] fp32
  bf16_t* feats = (bf16_t*)bufB;   // [1200,1536]
  float*  gates = (float*)bufA;    // [1200,1024] fp32
  bf16_t* zbf   = (bf16_t*)bufB;   // [1200,256]
  float*  logits= (float*)bufA;    // [1200,500]

  float* probs = out;              // [16,75,500]
  float* z     = out + 600000;     // [16,75,256]
  float* hN    = out + 907200;     // [1,16,256]
  float* cN    = out + 911296;     // [1,16,256]

  // ---- weight conversions --------------------------------------------------
  cvt_tap_kernel<<<(27 * 64 * 32 + 255) / 256, 256, 0, stream>>>(conv2_w, wt2, 64, 32);
  cvt_tap_kernel<<<(27 * 96 * 64 + 255) / 256, 256, 0, stream>>>(conv3_w, wt3, 96, 64);
  f2bf_kernel<<<(2048 * 3072 + 255) / 256, 256, 0, stream>>>(lin1_w, l1w, 2048 * 3072);
  f2bf_kernel<<<(512 * 2048 + 255) / 256, 256, 0, stream>>>(lin2_w, l2w, 512 * 2048);
  f2bf_kernel<<<(1024 * 1536 + 255) / 256, 256, 0, stream>>>(w_ih, wihb, 1024 * 1536);
  f2bf_kernel<<<(1024 * 256 + 255) / 256, 256, 0, stream>>>(w_hh, whhb, 1024 * 256);
  f2bf_kernel<<<(500 * 256 + 255) / 256, 256, 0, stream>>>(cls_w, clswb, 500 * 256);

  // ---- conv encoder --------------------------------------------------------
  conv1_fused_kernel<<<76800, 256, 0, stream>>>(x, conv1_w, conv1_b, act1);
  conv3d_gemm_kernel<<<dim3(11475, 1), 256, 0, stream>>>(act1, wt2, conv2_b, act2,
                                                         734400, 16, 32, 32, 18, 34, 64, 2);
  pool2_kernel<<<45900, 256, 0, stream>>>(act2, act2p);
  conv3d_gemm_kernel<<<dim3(2869, 2), 256, 0, stream>>>(act2p, wt3, conv3_b, act3,
                                                        183600, 9, 17, 64, 9, 17, 96, 1);
  pool3_flat_kernel<<<14400, 256, 0, stream>>>(act3, xflat);

  // ---- linear encoder ------------------------------------------------------
  gemm_bf16_kernel<<<dim3(19, 16), 256, 0, stream>>>(xflat, l1w, lin1_b, nullptr,
                                                     nullptr, a1, 1200, 2048, 3072, 1);
  gemm_bf16_kernel<<<dim3(19, 4), 256, 0, stream>>>(a1, l2w, lin2_b, nullptr,
                                                    x512, nullptr, 1200, 512, 2048, 1);

  // ---- delta features ------------------------------------------------------
  delta_feats_kernel<<<2400, 256, 0, stream>>>(x512, feats);

  // ---- LSTM ----------------------------------------------------------------
  gemm_bf16_kernel<<<dim3(19, 8), 256, 0, stream>>>(feats, wihb, b_ih, b_hh,
                                                    gates, nullptr, 1200, 1024, 1536, 0);
  lstm_kernel<<<1, 1024, 73728, stream>>>(gates, whhb, h0, c0, z, hN, cN);

  // ---- classifier + softmax ------------------------------------------------
  f2bf_kernel<<<1200, 256, 0, stream>>>(z, zbf, 307200);
  gemm_bf16_kernel<<<dim3(19, 4), 256, 0, stream>>>(zbf, clswb, cls_b, nullptr,
                                                    logits, nullptr, 1200, 500, 256, 0);
  softmax_kernel<<<150, 256, 0, stream>>>(logits, probs);
}
